// LSTM_2293512536657
// MI455X (gfx1250) — matrix operand order repaired and checked by reading, not compiled, not run on hardware
//
#include <hip/hip_runtime.h>

#define Bdim 256
#define Tdim 256
#define INd  2
#define Hd   1024
#define H4d  4096

typedef __attribute__((ext_vector_type(16))) __bf16 v16bf;
typedef __attribute__((ext_vector_type(8)))  __bf16 v8bf;
typedef __attribute__((ext_vector_type(8)))  float  v8f;

static __device__ __forceinline__ float sigmoidf_(float x) {
    return 1.0f / (1.0f + __expf(-x));
}
static __device__ __forceinline__ float tanhf_(float x) {
    float e = __expf(-2.0f * fabsf(x));   // e in (0,1], no overflow
    float t = (1.0f - e) / (1.0f + e);
    return copysignf(t, x);
}

// ---------------------------------------------------------------------------
// One-time pack of Uz (fp32, row-major H x 4H) into the WMMA B-operand lane
// layout, split into bf16 hi + lo parts. Packed layout:
//   packed[(((cTile*32)+q)*32 + lane)*16 + e] = Uz[k][n]
//   n = cTile*16 + (lane&15),  k = q*32 + (lane>>4)*16 + e
// so each lane's 16 bf16 B-fragment is one contiguous 32-byte load.
// ---------------------------------------------------------------------------
__global__ void pack_uz_kernel(const float* __restrict__ Uz,
                               __bf16* __restrict__ hiP,
                               __bf16* __restrict__ loP) {
    int tid  = blockIdx.x * blockDim.x + threadIdx.x;  // 0 .. 262143
    int lane = tid & 31;
    int q    = (tid >> 5) & 31;
    int c    = tid >> 10;                               // col tile 0..255 (over 4096)
    int n    = c * 16 + (lane & 15);
    int kb   = q * 32 + (lane >> 4) * 16;
    v16bf hv, lv;
#pragma unroll
    for (int e = 0; e < 16; ++e) {
        float v  = Uz[(size_t)(kb + e) * H4d + n];
        __bf16 h = (__bf16)v;
        hv[e] = h;
        lv[e] = (__bf16)(v - (float)h);
    }
    size_t base = (size_t)tid * 16;
    *(v16bf*)(hiP + base) = hv;
    *(v16bf*)(loP + base) = lv;
}

__global__ void init_h_kernel(const float* __restrict__ h0,
                              float* __restrict__ hF,
                              __bf16* __restrict__ hHi,
                              __bf16* __restrict__ hLo) {
    int i = blockIdx.x * blockDim.x + threadIdx.x;   // 0 .. B*H-1
    float v = h0[i];
    hF[i] = v;
    __bf16 hb = (__bf16)v;
    hHi[i] = hb;
    hLo[i] = (__bf16)(v - (float)hb);
}

// outputs[b,t,:] = x_t + tau * (x_t @ A^T); A = [[1.5,-1.5],[1/1.5,0]]
__global__ void outputs_kernel(const float* __restrict__ x,
                               const float* __restrict__ tau,
                               float* __restrict__ out) {
    int i = blockIdx.x * blockDim.x + threadIdx.x;   // i = b*T + t
    int b = i >> 8;                                   // T = 256
    float x0 = x[(size_t)i * 2 + 0];
    float x1 = x[(size_t)i * 2 + 1];
    out[(size_t)i * 2 + 0] = x0 + tau[b * 2 + 0] * (1.5f * x0 - 1.5f * x1);
    out[(size_t)i * 2 + 1] = x1 + tau[b * 2 + 1] * (x0 * (1.0f / 1.5f));
}

// ---------------------------------------------------------------------------
// One LSTM timestep. Grid: (64 hidden-col tiles, 16 batch-row tiles),
// block = 128 threads = 4 waves; wave w computes gate w's 16x16 tile of
// h @ Uz with v_wmma_f32_16x16x32_bf16 (bf16x2 split: hi*hi+hi*lo+lo*hi).
// ---------------------------------------------------------------------------
__global__ __launch_bounds__(128)
void lstm_step_kernel(const __bf16* __restrict__ uzHi, const __bf16* __restrict__ uzLo,
                      const float*  __restrict__ hPrev,
                      const __bf16* __restrict__ hHi,  const __bf16* __restrict__ hLo,
                      float* __restrict__ hNext,
                      __bf16* __restrict__ hNextHi, __bf16* __restrict__ hNextLo,
                      const float* __restrict__ x,  const float* __restrict__ Wx,
                      const float* __restrict__ bx,
                      float* __restrict__ hOut, float* __restrict__ cOut, int t) {
    __shared__ float gsh[4][16][16];

    const int wave    = threadIdx.x >> 5;    // gate index 0..3 (i,f,g,o)
    const int lane    = threadIdx.x & 31;
    const int half    = lane >> 4;
    const int mn      = lane & 15;
    const int hidTile = blockIdx.x;          // 0..63
    const int rowBase = blockIdx.y * 16;     // batch row base
    const int cTile   = wave * 64 + hidTile; // col tile over 4096/16

    // A operand rows (one batch row per lane-pair half, same row both halves)
    const __bf16* aHiRow = hHi + (size_t)(rowBase + mn) * Hd;
    const __bf16* aLoRow = hLo + (size_t)(rowBase + mn) * Hd;
    const size_t  bBase  = ((size_t)cTile * 32 * 32 + lane) * 16;

    v8f acc = {};
    for (int q = 0; q < 32; ++q) {
        // A fragment (contiguous K, matches packed B): lanes 0-15: K q*32+0..15,
        // lanes 16-31: K q*32+16..31  -> two 16-byte loads
        const int kb = q * 32 + half * 16;
        union { v16bf v; v8bf h[2]; } aHi, aLo;
        aHi.h[0] = *(const v8bf*)(aHiRow + kb);
        aHi.h[1] = *(const v8bf*)(aHiRow + kb + 8);
        aLo.h[0] = *(const v8bf*)(aLoRow + kb);
        aLo.h[1] = *(const v8bf*)(aLoRow + kb + 8);
        // B fragment: one contiguous 32-byte load from the packed layout
        v16bf bHi = *(const v16bf*)(uzHi + bBase + (size_t)q * 512);
        v16bf bLo = *(const v16bf*)(uzLo + bBase + (size_t)q * 512);

        acc = __builtin_amdgcn_wmma_f32_16x16x32_bf16(false, aHi.v, false, bHi,
                                                      (short)0, acc, false, false);
        acc = __builtin_amdgcn_wmma_f32_16x16x32_bf16(false, aHi.v, false, bLo,
                                                      (short)0, acc, false, false);
        acc = __builtin_amdgcn_wmma_f32_16x16x32_bf16(false, aLo.v, false, bHi,
                                                      (short)0, acc, false, false);
    }

    // C/D layout: VGPR r -> (M = r + 8*half, N = lane&15)
#pragma unroll
    for (int r = 0; r < 8; ++r)
        gsh[wave][r + half * 8][mn] = acc[r];
    __syncthreads();

    // Fused elementwise phase: 128 threads cover the 256-element tile
    for (int idx = threadIdx.x; idx < 256; idx += 128) {
        int m    = idx >> 4;
        int n    = idx & 15;
        int b    = rowBase + m;
        int hcol = hidTile * 16 + n;
        float x0 = x[((size_t)b * Tdim + t) * 2 + 0];
        float x1 = x[((size_t)b * Tdim + t) * 2 + 1];
        float g4[4];
#pragma unroll
        for (int gidx = 0; gidx < 4; ++gidx) {
            int gcol = gidx * Hd + hcol;
            g4[gidx] = gsh[gidx][m][n] + x0 * Wx[gcol] + x1 * Wx[H4d + gcol] + bx[gcol];
        }
        float gi = sigmoidf_(g4[0]);
        float gf = sigmoidf_(g4[1]);
        float gg = tanhf_(g4[2]);
        float go = sigmoidf_(g4[3]);
        size_t o = (size_t)b * Hd + hcol;
        float hp = hPrev[o];
        float c  = gf * hp + gi * gg;       // faithful: uses h, not c
        float h  = go * tanhf_(c);
        hNext[o] = h;
        __bf16 hb = (__bf16)h;
        hNextHi[o] = hb;
        hNextLo[o] = (__bf16)(h - (float)hb);
        hOut[o] = h;                        // hT region (last step wins)
        cOut[o] = c;                        // cT region (last step wins)
    }
}

extern "C" void kernel_launch(void* const* d_in, const int* in_sizes, int n_in,
                              void* d_out, int out_size, void* d_ws, size_t ws_size,
                              hipStream_t stream) {
    (void)in_sizes; (void)n_in; (void)out_size; (void)ws_size;
    // setup_inputs order: x, h0, c0, u, tau, Wx, bx, Uz  (c0 and u are dead)
    const float* x   = (const float*)d_in[0];
    const float* h0  = (const float*)d_in[1];
    const float* tau = (const float*)d_in[4];
    const float* Wx  = (const float*)d_in[5];
    const float* bx  = (const float*)d_in[6];
    const float* Uz  = (const float*)d_in[7];

    float* out = (float*)d_out;
    float* hT  = out + (size_t)Bdim * Tdim * INd;   // + 131072
    float* cT  = hT + (size_t)Bdim * Hd;            // + 262144

    // Workspace layout (~20 MB): Uz hi/lo packs + ping-pong h (f32, bf16 hi/lo)
    const size_t nUz = (size_t)Hd * H4d;            // 4194304
    const size_t nH  = (size_t)Bdim * Hd;           // 262144
    char* ws = (char*)d_ws;
    __bf16* uzHi = (__bf16*)ws;  ws += nUz * 2;
    __bf16* uzLo = (__bf16*)ws;  ws += nUz * 2;
    float*  hF[2];  __bf16* hHi[2]; __bf16* hLo[2];
    hF[0]  = (float*)ws;   ws += nH * 4;
    hF[1]  = (float*)ws;   ws += nH * 4;
    hHi[0] = (__bf16*)ws;  ws += nH * 2;
    hHi[1] = (__bf16*)ws;  ws += nH * 2;
    hLo[0] = (__bf16*)ws;  ws += nH * 2;
    hLo[1] = (__bf16*)ws;  ws += nH * 2;

    pack_uz_kernel<<<1024, 256, 0, stream>>>(Uz, uzHi, uzLo);
    init_h_kernel<<<1024, 256, 0, stream>>>(h0, hF[0], hHi[0], hLo[0]);
    outputs_kernel<<<256, 256, 0, stream>>>(x, tau, out);

    for (int t = 0; t < Tdim; ++t) {
        int a = t & 1, b = a ^ 1;
        lstm_step_kernel<<<dim3(64, 16), 128, 0, stream>>>(
            uzHi, uzLo,
            hF[a], hHi[a], hLo[a],
            hF[b], hHi[b], hLo[b],
            x, Wx, bx, hT, cT, t);
    }
}
